// BigBirdAttention_66803921322326
// MI455X (gfx1250) — matrix operand order repaired and checked by reading, not compiled, not run on hardware
//
#include <hip/hip_runtime.h>
#include <hip/hip_bf16.h>
#include <cstdint>
#include <cstddef>

// ---------------------------------------------------------------------------
// BigBird sparse attention for MI455X (gfx1250, wave32, WMMA bf16 16x16x32)
// ---------------------------------------------------------------------------
#define D_MODEL 1024
#define NHEAD   16
#define DKH     64        // head dim (DK == DV)
#define BS_LOC  128       // local block size
#define BATCH   2
#define SEQ     4098
#define NBLK    32        // (SEQ-2)/BS_LOC
#define MTOT    (BATCH * SEQ)   // 8196 rows

typedef __attribute__((ext_vector_type(16))) __bf16 v16bf;
typedef __attribute__((ext_vector_type(8)))  float  v8f;

union FragB16 { uint4 q[2]; v16bf v; };

__device__ __forceinline__ unsigned short f2bf(float f) {
  union { float f; unsigned u; } c; c.f = f;
  unsigned u = c.u;
  u += 0x7FFFu + ((u >> 16) & 1u);          // round-to-nearest-even
  return (unsigned short)(u >> 16);
}
__device__ __forceinline__ float bf2f(unsigned short h) {
  union { unsigned u; float f; } c; c.u = ((unsigned)h) << 16;
  return c.f;
}

// gfx1250 async global->LDS copy (ASYNCcnt-tracked, no VGPR data round-trip).
// lds_dst is the 32-bit LDS byte address (low 32 bits of the generic pointer).
__device__ __forceinline__ void async_copy_b128(unsigned lds_dst, const void* gsrc) {
  asm volatile("global_load_async_to_lds_b128 %0, %1, off"
               :: "v"(lds_dst), "v"((unsigned long long)(uintptr_t)gsrc)
               : "memory");
}
__device__ __forceinline__ void async_wait0() {
#if __has_builtin(__builtin_amdgcn_s_wait_asynccnt)
  __builtin_amdgcn_s_wait_asynccnt(0);
#else
  asm volatile("s_wait_asynccnt 0x0" ::: "memory");
#endif
}
__device__ __forceinline__ unsigned lds_addr32(const void* p) {
  return (unsigned)(uintptr_t)p;
}

// ---------------------------------------------------------------------------
// f32 -> bf16 conversion
// ---------------------------------------------------------------------------
__global__ void __launch_bounds__(256)
bb_cvt_f32_to_bf16(const float* __restrict__ src,
                   unsigned short* __restrict__ dst, int n) {
  int i = blockIdx.x * blockDim.x + threadIdx.x;
  if (i < n) dst[i] = f2bf(src[i]);
}

// ---------------------------------------------------------------------------
// Generic bf16 WMMA GEMM:  C[M,N] = outScale * (A[M,K] * Bt[N,K]^T)
// outF32 == 0 : C stored bf16 ; outF32 == 1 : C stored f32, bias added.
// 256 threads = 8 waves; wave tile 32x64; workgroup tile 64x256.
// ---------------------------------------------------------------------------
__global__ void __launch_bounds__(256)
bb_gemm_bf16(const unsigned short* __restrict__ A,
             const unsigned short* __restrict__ Bt,
             void* __restrict__ Cout,
             const float* __restrict__ bias,
             int M, int N, int K, int outF32, float outScale)
{
  const int lane  = threadIdx.x & 31;
  const int wid   = threadIdx.x >> 5;
  const int waveM = wid & 1;
  const int waveN = wid >> 1;
  const int half  = lane >> 4;
  const int l15   = lane & 15;
  const int n0 = blockIdx.x * 256 + waveN * 64;
  const int m0 = blockIdx.y * 64  + waveM * 32;

  const v8f vzero = {};
  v8f acc[2][4];
#pragma unroll
  for (int s = 0; s < 2; ++s)
#pragma unroll
    for (int j = 0; j < 4; ++j) acc[s][j] = vzero;

  int mA0 = m0 + l15;       if (mA0 >= M) mA0 = M - 1;   // clamp loads, mask stores
  int mA1 = m0 + 16 + l15;  if (mA1 >= M) mA1 = M - 1;
  const unsigned short* Ar0 = A + (size_t)mA0 * K;
  const unsigned short* Ar1 = A + (size_t)mA1 * K;
  const unsigned short* Br0 = Bt + (size_t)(n0 +  0 + l15) * K;
  const unsigned short* Br1 = Bt + (size_t)(n0 + 16 + l15) * K;
  const unsigned short* Br2 = Bt + (size_t)(n0 + 32 + l15) * K;
  const unsigned short* Br3 = Bt + (size_t)(n0 + 48 + l15) * K;

  for (int k0 = 0; k0 < K; k0 += 32) {
    // Prefetch the streamed A rows one K-block ahead, locality 3 -> WGP scope
    // so the line is pulled into the near cache (B rows are weight data shared
    // by every workgroup and stay resident in the 192 MB L2).
    __builtin_prefetch(Ar0 + k0 + 32, 0, 3);
    __builtin_prefetch(Ar1 + k0 + 32, 0, 3);

    FragB16 a0, a1, b0, b1, b2, b3;
    const int ka = k0 + half * 8;          // A frag: K = ka..ka+7, ka+16..ka+23
    a0.q[0] = *(const uint4*)(Ar0 + ka);
    a0.q[1] = *(const uint4*)(Ar0 + ka + 16);
    a1.q[0] = *(const uint4*)(Ar1 + ka);
    a1.q[1] = *(const uint4*)(Ar1 + ka + 16);
    const int kb = k0 + half * 8;          // B frag: K = kb..kb+7, kb+16..kb+23
    b0.q[0] = *(const uint4*)(Br0 + kb); b0.q[1] = *(const uint4*)(Br0 + kb + 16);
    b1.q[0] = *(const uint4*)(Br1 + kb); b1.q[1] = *(const uint4*)(Br1 + kb + 16);
    b2.q[0] = *(const uint4*)(Br2 + kb); b2.q[1] = *(const uint4*)(Br2 + kb + 16);
    b3.q[0] = *(const uint4*)(Br3 + kb); b3.q[1] = *(const uint4*)(Br3 + kb + 16);

    acc[0][0] = __builtin_amdgcn_wmma_f32_16x16x32_bf16(false, a0.v, false, b0.v, (short)0, acc[0][0], false, false);
    acc[0][1] = __builtin_amdgcn_wmma_f32_16x16x32_bf16(false, a0.v, false, b1.v, (short)0, acc[0][1], false, false);
    acc[0][2] = __builtin_amdgcn_wmma_f32_16x16x32_bf16(false, a0.v, false, b2.v, (short)0, acc[0][2], false, false);
    acc[0][3] = __builtin_amdgcn_wmma_f32_16x16x32_bf16(false, a0.v, false, b3.v, (short)0, acc[0][3], false, false);
    acc[1][0] = __builtin_amdgcn_wmma_f32_16x16x32_bf16(false, a1.v, false, b0.v, (short)0, acc[1][0], false, false);
    acc[1][1] = __builtin_amdgcn_wmma_f32_16x16x32_bf16(false, a1.v, false, b1.v, (short)0, acc[1][1], false, false);
    acc[1][2] = __builtin_amdgcn_wmma_f32_16x16x32_bf16(false, a1.v, false, b2.v, (short)0, acc[1][2], false, false);
    acc[1][3] = __builtin_amdgcn_wmma_f32_16x16x32_bf16(false, a1.v, false, b3.v, (short)0, acc[1][3], false, false);
  }

  // C element (m = base + i + 8*half, n = n0 + 16j + l15)
#pragma unroll
  for (int s = 0; s < 2; ++s) {
#pragma unroll
    for (int j = 0; j < 4; ++j) {
      const int n = n0 + j * 16 + l15;
      const float bv = (outF32 && bias) ? bias[n] : 0.f;
#pragma unroll
      for (int i = 0; i < 8; ++i) {
        const int m = m0 + s * 16 + i + 8 * half;
        if (m < M) {
          const float v = acc[s][j][i] * outScale;
          if (outF32) ((float*)Cout)[(size_t)m * N + n] = v + bv;
          else ((unsigned short*)Cout)[(size_t)m * N + n] = f2bf(v);
        }
      }
    }
  }
}

// ---------------------------------------------------------------------------
// Local 3-block attention + global-token term (flash online softmax).
// grid = ((SEQ-2)/64, H, B); 128 threads = 4 waves; wave owns 16 query rows.
// Q is pre-scaled by 1/sqrt(dk) in the Q-projection GEMM epilogue, so both Q
// and K tiles are staged with gfx1250 async global->LDS copies.
// ---------------------------------------------------------------------------
__global__ void __launch_bounds__(128)
bb_local_attn(const unsigned short* __restrict__ Q,
              const unsigned short* __restrict__ K,
              const unsigned short* __restrict__ V,
              unsigned short* __restrict__ O)
{
  __shared__ unsigned short sQ[64 * DKH];          //  8 KB (pre-scaled Q)
  __shared__ unsigned short sK[BS_LOC * DKH];      // 16 KB key chunk, row-major
  __shared__ unsigned short sVt[DKH * BS_LOC];     // 16 KB value chunk, transposed
  __shared__ unsigned short sP[4][16 * BS_LOC];    // 16 KB per-wave P strip
  __shared__ float          sWg[64][2];            // global-token softmax weights

  const int tid  = threadIdx.x;
  const int lane = tid & 31;
  const int wid  = tid >> 5;
  const int half = lane >> 4;
  const int l15  = lane & 15;

  const int rg   = blockIdx.x;          // 64-row group
  const int h    = blockIdx.y;
  const int b    = blockIdx.z;
  const int nblk = rg >> 1;             // 128-token block index
  const int t0   = 1 + rg * 64;         // first global token of this row group
  const size_t headOff = (size_t)h * DKH;

  const int j8 = tid >> 3;              // 0..15
  const int d8 = (tid & 7) * 8;         // 0..56, 16B granules

  // ---- stage Q asynchronously (completion covered by first chunk's wait) ----
#pragma unroll
  for (int it = 0; it < 4; ++it) {
    const int r = it * 16 + j8;
    async_copy_b128(lds_addr32(&sQ[r * DKH + d8]),
                    Q + ((size_t)(b * SEQ + t0 + r)) * D_MODEL + headOff + d8);
  }

  float mrow[8], lrow[8];
#pragma unroll
  for (int i = 0; i < 8; ++i) { mrow[i] = -1e30f; lrow[i] = 0.f; }
  const v8f vzero = {};
  v8f o[4];
#pragma unroll
  for (int t = 0; t < 4; ++t) o[t] = vzero;

  const int m0 = wid * 16;              // wave's row strip within the 64 rows

  for (int c = 0; c < 3; ++c) {
    const int kb = nblk - 1 + c;        // key block; skipped == reference -inf mask
    if (kb < 0 || kb >= NBLK) continue; // uniform per workgroup
    __syncthreads();                    // protect sK/sVt/sP reuse

    // ---- stage K chunk (async, row-major) and V chunk (transposed) ----
    {
      const int kt0 = 1 + kb * BS_LOC;
#pragma unroll
      for (int it = 0; it < 8; ++it) {
        const int r = it * 16 + j8;
        const size_t g = ((size_t)(b * SEQ + kt0 + r)) * D_MODEL + headOff + d8;
        async_copy_b128(lds_addr32(&sK[r * DKH + d8]), K + g);
        uint4 vv = *(const uint4*)(V + g);
        const unsigned short* pv = (const unsigned short*)&vv;
#pragma unroll
        for (int e = 0; e < 8; ++e) sVt[(d8 + e) * BS_LOC + r] = pv[e];
      }
    }
    async_wait0();                      // also covers the Q tile on chunk 0
    __syncthreads();

    // ---- S = Q K^T for this wave's 16x128 strip ----
    FragB16 aQ[2];
#pragma unroll
    for (int ks = 0; ks < 2; ++ks) {
      const int ab = (m0 + l15) * DKH + ks * 32 + half * 8;
      aQ[ks].q[0] = *(const uint4*)(&sQ[ab]);
      aQ[ks].q[1] = *(const uint4*)(&sQ[ab + 16]);
    }
    v8f s[8];
#pragma unroll
    for (int t = 0; t < 8; ++t) {
      v8f a = vzero;
#pragma unroll
      for (int ks = 0; ks < 2; ++ks) {
        FragB16 bb;
        const int bbse = (t * 16 + l15) * DKH + ks * 32 + half * 8;
        bb.q[0] = *(const uint4*)(&sK[bbse]);
        bb.q[1] = *(const uint4*)(&sK[bbse + 16]);
        a = __builtin_amdgcn_wmma_f32_16x16x32_bf16(false, aQ[ks].v, false, bb.v,
                                                    (short)0, a, false, false);
      }
      s[t] = a;
    }

    // ---- online softmax (row m = i + 8*half within strip) ----
    float cmax[8];
#pragma unroll
    for (int i = 0; i < 8; ++i) {
      float m = s[0][i];
#pragma unroll
      for (int t = 1; t < 8; ++t) m = fmaxf(m, s[t][i]);
      cmax[i] = m;
    }
#pragma unroll
    for (int off = 1; off < 16; off <<= 1)
#pragma unroll
      for (int i = 0; i < 8; ++i)
        cmax[i] = fmaxf(cmax[i], __shfl_xor(cmax[i], off, 32));

    float scl[8], psum[8];
#pragma unroll
    for (int i = 0; i < 8; ++i) {
      const float mn = fmaxf(mrow[i], cmax[i]);
      scl[i]  = __expf(mrow[i] - mn);
      mrow[i] = mn;
      psum[i] = 0.f;
    }
#pragma unroll
    for (int t = 0; t < 8; ++t)
#pragma unroll
      for (int i = 0; i < 8; ++i) {
        const float p = __expf(s[t][i] - mrow[i]);
        psum[i] += p;
        sP[wid][(i + 8 * half) * BS_LOC + t * 16 + l15] = f2bf(p);
      }
#pragma unroll
    for (int off = 1; off < 16; off <<= 1)
#pragma unroll
      for (int i = 0; i < 8; ++i) psum[i] += __shfl_xor(psum[i], off, 32);
#pragma unroll
    for (int i = 0; i < 8; ++i) lrow[i] = lrow[i] * scl[i] + psum[i];
#pragma unroll
    for (int t = 0; t < 4; ++t)
#pragma unroll
      for (int i = 0; i < 8; ++i) o[t][i] *= scl[i];

    // ---- O += P * V (contraction over the 128 chunk keys) ----
#pragma unroll
    for (int dt = 0; dt < 4; ++dt) {
#pragma unroll
      for (int ks = 0; ks < 4; ++ks) {
        FragB16 a, bb;
        const int ab = l15 * BS_LOC + ks * 32 + half * 8;
        a.q[0] = *(const uint4*)(&sP[wid][ab]);
        a.q[1] = *(const uint4*)(&sP[wid][ab + 16]);
        const int bbse = (dt * 16 + l15) * BS_LOC + ks * 32 + half * 8;
        bb.q[0] = *(const uint4*)(&sVt[bbse]);
        bb.q[1] = *(const uint4*)(&sVt[bbse + 16]);
        o[dt] = __builtin_amdgcn_wmma_f32_16x16x32_bf16(false, a.v, false, bb.v,
                                                        (short)0, o[dt], false, false);
      }
    }
  }

  // ---- separately-softmaxed global-token (2 keys) contribution ----
  __syncthreads();
  if (tid < 64) {
    const unsigned short* kg0 = K + ((size_t)(b * SEQ + 0))        * D_MODEL + headOff;
    const unsigned short* kg1 = K + ((size_t)(b * SEQ + SEQ - 1))  * D_MODEL + headOff;
    float s0 = 0.f, s1 = 0.f;
#pragma unroll
    for (int d = 0; d < DKH; ++d) {
      const float qd = bf2f(sQ[tid * DKH + d]);   // already scaled by 1/sqrt(dk)
      s0 += qd * bf2f(kg0[d]);
      s1 += qd * bf2f(kg1[d]);
    }
    const float mg = fmaxf(s0, s1);
    const float e0 = __expf(s0 - mg), e1 = __expf(s1 - mg);
    const float inv = 1.f / (e0 + e1);
    sWg[tid][0] = e0 * inv;
    sWg[tid][1] = e1 * inv;
  }
  __syncthreads();

  const unsigned short* vg0 = V + ((size_t)(b * SEQ + 0))       * D_MODEL + headOff;
  const unsigned short* vg1 = V + ((size_t)(b * SEQ + SEQ - 1)) * D_MODEL + headOff;
#pragma unroll
  for (int dt = 0; dt < 4; ++dt) {
    const int dv = dt * 16 + l15;
    const float g0 = bf2f(vg0[dv]);
    const float g1 = bf2f(vg1[dv]);
#pragma unroll
    for (int i = 0; i < 8; ++i) {
      const int rloc = wid * 16 + i + 8 * half;
      const float val = o[dt][i] / lrow[i] + sWg[rloc][0] * g0 + sWg[rloc][1] * g1;
      O[((size_t)(b * SEQ + t0 + rloc)) * D_MODEL + headOff + dv] = f2bf(val);
    }
  }
}

// ---------------------------------------------------------------------------
// Global-query attention: 2 queries per (b,h) over all SEQ keys. Tiny.
// grid = (2, H, B); 256 threads.  NOTE: Q is pre-scaled by 1/sqrt(dk).
// ---------------------------------------------------------------------------
__global__ void __launch_bounds__(256)
bb_global_attn(const unsigned short* __restrict__ Q,
               const unsigned short* __restrict__ K,
               const unsigned short* __restrict__ V,
               unsigned short* __restrict__ O)
{
  __shared__ float sS[SEQ + 14];
  __shared__ float sq[DKH];
  __shared__ float sred[256];

  const int g = blockIdx.x, h = blockIdx.y, b = blockIdx.z;
  const int tq = (g == 0) ? 0 : (SEQ - 1);
  const int tid = threadIdx.x;
  const size_t headOff = (size_t)h * DKH;

  if (tid < DKH)
    sq[tid] = bf2f(Q[((size_t)(b * SEQ + tq)) * D_MODEL + headOff + tid]);
  __syncthreads();

  float lmax = -1e30f;
  for (int t = tid; t < SEQ; t += 256) {
    const unsigned short* kr = K + ((size_t)(b * SEQ + t)) * D_MODEL + headOff;
    float s = 0.f;
#pragma unroll
    for (int d = 0; d < DKH; ++d) s += sq[d] * bf2f(kr[d]);
    sS[t] = s;
    lmax = fmaxf(lmax, s);
  }
  sred[tid] = lmax;
  __syncthreads();
  for (int off = 128; off > 0; off >>= 1) {
    if (tid < off) sred[tid] = fmaxf(sred[tid], sred[tid + off]);
    __syncthreads();
  }
  const float gmax = sred[0];
  __syncthreads();

  float lsum = 0.f;
  for (int t = tid; t < SEQ; t += 256) {
    const float p = __expf(sS[t] - gmax);
    sS[t] = p;
    lsum += p;
  }
  sred[tid] = lsum;
  __syncthreads();
  for (int off = 128; off > 0; off >>= 1) {
    if (tid < off) sred[tid] += sred[tid + off];
    __syncthreads();
  }
  const float inv = 1.f / sred[0];
  __syncthreads();

  const int dv = tid & 63, grp = tid >> 6;
  float acc = 0.f;
  for (int t = grp; t < SEQ; t += 4)
    acc += sS[t] * bf2f(V[((size_t)(b * SEQ + t)) * D_MODEL + headOff + dv]);
  sred[tid] = acc;
  __syncthreads();
  if (grp == 0) {
    const float tot = sred[dv] + sred[64 + dv] + sred[128 + dv] + sred[192 + dv];
    O[((size_t)(b * SEQ + tq)) * D_MODEL + headOff + dv] = f2bf(tot * inv);
  }
}

// ---------------------------------------------------------------------------
// Host-side orchestration (all on `stream`, graph-capture safe)
// ---------------------------------------------------------------------------
extern "C" void kernel_launch(void* const* d_in, const int* in_sizes, int n_in,
                              void* d_out, int out_size, void* d_ws, size_t ws_size,
                              hipStream_t stream) {
  const float* x  = (const float*)d_in[0];
  const float* Wq = (const float*)d_in[1];
  const float* Wk = (const float*)d_in[2];
  const float* Wv = (const float*)d_in[3];
  const float* Wo = (const float*)d_in[4];
  const float* bo = (const float*)d_in[5];

  const size_t nX = (size_t)MTOT * D_MODEL;       // 8,392,704 elems
  const size_t nW = (size_t)D_MODEL * D_MODEL;    // 1,048,576 elems

  unsigned short* ws  = (unsigned short*)d_ws;
  unsigned short* xb  = ws;            // bf16 x
  unsigned short* Wqb = xb  + nX;
  unsigned short* Wkb = Wqb + nW;
  unsigned short* Wvb = Wkb + nW;
  unsigned short* Wob = Wvb + nW;
  unsigned short* Qb  = Wob + nW;      // bf16 Q/K/V, (B*T, H*64) row-major
  unsigned short* Kb  = Qb  + nX;
  unsigned short* Vb  = Kb  + nX;
  unsigned short* Ab  = Vb  + nX;      // bf16 attention output (B*T, H*64)

  bb_cvt_f32_to_bf16<<<dim3((unsigned)((nX + 255) / 256)), dim3(256), 0, stream>>>(x,  xb,  (int)nX);
  bb_cvt_f32_to_bf16<<<dim3((unsigned)((nW + 255) / 256)), dim3(256), 0, stream>>>(Wq, Wqb, (int)nW);
  bb_cvt_f32_to_bf16<<<dim3((unsigned)((nW + 255) / 256)), dim3(256), 0, stream>>>(Wk, Wkb, (int)nW);
  bb_cvt_f32_to_bf16<<<dim3((unsigned)((nW + 255) / 256)), dim3(256), 0, stream>>>(Wv, Wvb, (int)nW);
  bb_cvt_f32_to_bf16<<<dim3((unsigned)((nW + 255) / 256)), dim3(256), 0, stream>>>(Wo, Wob, (int)nW);

  dim3 gg(D_MODEL / 256, (MTOT + 63) / 64);       // (4, 129)
  // Q projection folds the 1/sqrt(dk) attention scale into its epilogue.
  bb_gemm_bf16<<<gg, dim3(256), 0, stream>>>(xb, Wqb, (void*)Qb, nullptr, MTOT, D_MODEL, D_MODEL, 0, 0.125f);
  bb_gemm_bf16<<<gg, dim3(256), 0, stream>>>(xb, Wkb, (void*)Kb, nullptr, MTOT, D_MODEL, D_MODEL, 0, 1.0f);
  bb_gemm_bf16<<<gg, dim3(256), 0, stream>>>(xb, Wvb, (void*)Vb, nullptr, MTOT, D_MODEL, D_MODEL, 0, 1.0f);

  dim3 ga((SEQ - 2) / 64, NHEAD, BATCH);          // (64, 16, 2)
  bb_local_attn<<<ga, dim3(128), 0, stream>>>(Qb, Kb, Vb, Ab);

  dim3 gq(2, NHEAD, BATCH);
  bb_global_attn<<<gq, dim3(256), 0, stream>>>(Qb, Kb, Vb, Ab);

  bb_gemm_bf16<<<gg, dim3(256), 0, stream>>>(Ab, Wob, d_out, bo, MTOT, D_MODEL, D_MODEL, 1, 1.0f);

  (void)in_sizes; (void)n_in; (void)out_size; (void)ws_size;
}
